// RNN_20255065768518
// MI455X (gfx1250) — matrix operand order repaired and checked by reading, compiled, not run on hardware
//
#include <hip/hip_runtime.h>
#include <hip/hip_bf16.h>

// Problem constants (match the reference).
#define NN 64
#define LL 4096
#define CC 256
#define HH 256
#define PITCH 264   // LDS row pitch in bf16 elems: 264*2B = 528B = 132 dwords -> bank stride 4, conflict-free b128 reads

typedef __attribute__((ext_vector_type(16))) __bf16 v16bf;
typedef __attribute__((ext_vector_type(8)))  __bf16 v8bf;
typedef __attribute__((ext_vector_type(8)))  float  v8f;
typedef __attribute__((ext_vector_type(4)))  float  v4f;

// Load one 16x32 bf16 A-fragment from a 16xPITCH row-major LDS tile.
// ISA layout (16-bit A 16x32): lanes 0-15 -> row m=lane, K = kb+0..7 (V0-3), kb+16..23 (V4-7);
// lanes 16-31 -> row m=lane-16, K = kb+8..15 and kb+24..31. Both halves are 16B-aligned b128 reads.
__device__ __forceinline__ v16bf load_afrag(const __bf16* buf, int m, int kb) {
    const v8bf* p0 = reinterpret_cast<const v8bf*>(buf + m * PITCH + kb);
    const v8bf* p1 = reinterpret_cast<const v8bf*>(buf + m * PITCH + kb + 16);
    v8bf lo = *p0;
    v8bf hi = *p1;
    return __builtin_shufflevector(lo, hi, 0,1,2,3,4,5,6,7,8,9,10,11,12,13,14,15);
}

// gfx1250 has a hardware v_tanh_f32 TRANS op; use it on the recurrence critical path.
__device__ __forceinline__ float fast_tanh(float v) {
#if defined(__has_builtin)
#if __has_builtin(__builtin_amdgcn_tanhf)
    return __builtin_amdgcn_tanhf(v);
#else
    return tanhf(v);
#endif
#else
    return tanhf(v);
#endif
}

__global__ __launch_bounds__(512, 1)
void rnn_bwd_scan(const float* __restrict__ x,          // [N, L, C] f32
                  const long long* __restrict__ lengths, // [N] i64
                  const float* __restrict__ init_states, // [H] f32
                  const float* __restrict__ W_ih,        // [H, C] f32
                  const float* __restrict__ W_hh,        // [H, H] f32
                  const float* __restrict__ b_ih,        // [H]
                  const float* __restrict__ b_hh,        // [H]
                  float* __restrict__ out)               // [N, L, H] f32
{
    __shared__ __align__(16) __bf16 xbuf[2][16 * PITCH]; // staged reversed inputs (bf16)
    __shared__ __align__(16) __bf16 hbuf[2][16 * PITCH]; // recurrent state (bf16)

    const int tid  = threadIdx.x;
    const int lane = tid & 31;
    const int wave = tid >> 5;               // 16 waves
    const int i0   = blockIdx.x * 16;        // batch-row tile
    const int h0   = wave * 16;              // output-column tile for this wave
    const int col  = h0 + (lane & 15);       // C/D column for this lane
    const int moff = (lane >> 4) << 3;       // C/D rows: v + (lane<16 ? 0 : 8)
    const int kboff = (lane >> 4) << 3;      // A-frag K sub-offset (0 or 8)
    const int kwoff = (lane >> 4) << 3;      // B-frag K sub-offset (0 or 8)

    // ---- Persistent bf16 B-fragments for W_ih^T and W_hh^T tiles (held in VGPRs) ----
    // B[k][n] = W[h0+n][k]; lane (n = lane&15) holds K = kb+0..7 and kb+16..23 from row `col` of W.
    v16bf Bih[8], Bhh[8];
    {
        const float* wih_row = W_ih + (long long)col * CC;
        const float* whh_row = W_hh + (long long)col * HH;
#pragma unroll
        for (int c = 0; c < 8; ++c) {
            const int kb = c * 32 + kwoff;
#pragma unroll
            for (int j = 0; j < 16; ++j) {
                Bih[c][j] = (__bf16)wih_row[kb + j + ((j >> 3) << 3)];
                Bhh[c][j] = (__bf16)whh_row[kb + j + ((j >> 3) << 3)];
            }
        }
    }

    // ---- Per-lane output bookkeeping for its 8 C/D rows ----
    const float bias = b_ih[col] + b_hh[col];
    int       lenv[8];
    long long obase[8];
#pragma unroll
    for (int v = 0; v < 8; ++v) {
        const int m = moff + v;
        const long long lm = lengths[i0 + m];
        lenv[v]  = (int)lm;
        obase[v] = (((long long)(i0 + m)) * LL + (lm - 1)) * HH + col;  // unused when lm==0
    }

    // ---- Staging mapping: 32 threads per row, 8 columns per thread ----
    const int srow = tid >> 5;
    const int scol = (tid & 31) * 8;
    const int slen = (int)lengths[i0 + srow];
    const float* xrow = x + ((long long)(i0 + srow) * LL) * CC + scol;

    // h_{-1} = broadcast(init_states)  (bf16, into hbuf[0])
    {
        v8bf h0v;
#pragma unroll
        for (int j = 0; j < 8; ++j) h0v[j] = (__bf16)init_states[scol + j];
        *reinterpret_cast<v8bf*>(&hbuf[0][srow * PITCH + scol]) = h0v;
    }

    const int m = lane & 15;   // A-fragment row for this lane

    // ================= time loop (reversed frame): t = 0 .. L-1 =================
    for (int t = 0; t < LL; ++t) {
        const int pb = t & 1;
        const int pn = pb ^ 1;

        // ---- stage xr[i, t] = x[i, len-1-t] (zeros past length) into xbuf[pb] ----
        v8bf xb;
        if (t < slen) {
            const float* p = xrow + (long long)(slen - 1 - t) * CC;
            v4f f0 = reinterpret_cast<const v4f*>(p)[0];
            v4f f1 = reinterpret_cast<const v4f*>(p)[1];
#pragma unroll
            for (int j = 0; j < 4; ++j) { xb[j] = (__bf16)f0[j]; xb[4 + j] = (__bf16)f1[j]; }
            if (t + 4 < slen)  // pull a future step's row toward the caches (global_prefetch_b8)
                __builtin_prefetch(xrow + (long long)(slen - 5 - t) * CC, 0, 1);
        } else {
#pragma unroll
            for (int j = 0; j < 8; ++j) xb[j] = (__bf16)0.0f;
        }
        *reinterpret_cast<v8bf*>(&xbuf[pb][srow * PITCH + scol]) = xb;

        __syncthreads();   // xbuf[pb] staged; hbuf[pb] writes from previous step visible

        // ---- Z = bias + xr_t @ W_ih^T + h_{t-1} @ W_hh^T  (16x16 f32 tile per wave) ----
        // 16 WMMAs as one virtual sequence (k=0..7 -> xbuf x W_ih, k=8..15 -> hbuf x W_hh)
        // with a depth-2 software pipeline on the A-fragment LDS loads so ds latency
        // hides behind WMMA execution instead of forcing s_wait_dscnt 0 per WMMA.
        v8f acc;
#pragma unroll
        for (int v = 0; v < 8; ++v) acc[v] = bias;

        const __bf16* xb_base = &xbuf[pb][0];
        const __bf16* hb_base = &hbuf[pb][0];

#define SRC_FRAG(K) load_afrag(((K) < 8 ? xb_base : hb_base), m, ((K) & 7) * 32 + kboff)

        v16bf aA = SRC_FRAG(0);
        v16bf aB = SRC_FRAG(1);
#pragma unroll
        for (int k = 0; k < 16; ++k) {
            v16bf aN = (k + 2 < 16) ? SRC_FRAG(k + 2) : aA;
            const v16bf bfrag = (k < 8) ? Bih[k] : Bhh[k - 8];
            acc = __builtin_amdgcn_wmma_f32_16x16x32_bf16(false, aA, false, bfrag,
                                                          (short)0, acc, false, false);
            aA = aB;
            aB = aN;
        }
#undef SRC_FRAG

        // ---- h = tanh(Z): feed back via LDS (bf16), emit valid outputs (f32) ----
#pragma unroll
        for (int v = 0; v < 8; ++v) {
            const float hv = fast_tanh(acc[v]);
            hbuf[pn][(moff + v) * PITCH + col] = (__bf16)hv;
            if (t < lenv[v])
                out[obase[v] - (long long)t * HH] = hv;   // position s = len-1-t
        }
        // next iteration's __syncthreads separates these writes from the next reads
    }
}

// Zero out[i, s, :] for s >= len_i (reference zeroes past-length positions).
__global__ __launch_bounds__(64)
void zero_tail(float* __restrict__ out, const long long* __restrict__ lengths)
{
    const int b = blockIdx.x;          // b = i*L + s
    const int i = b >> 12;             // / 4096
    const int s = b & (LL - 1);
    if ((long long)s < lengths[i]) return;
    v4f z = {0.0f, 0.0f, 0.0f, 0.0f};
    reinterpret_cast<v4f*>(out)[(long long)b * (HH / 4) + threadIdx.x] = z;
}

extern "C" void kernel_launch(void* const* d_in, const int* in_sizes, int n_in,
                              void* d_out, int out_size, void* d_ws, size_t ws_size,
                              hipStream_t stream) {
    const float*     x      = (const float*)d_in[0];
    const long long* lens   = (const long long*)d_in[1];
    const float*     init_s = (const float*)d_in[2];
    const float*     W_ih   = (const float*)d_in[3];
    const float*     W_hh   = (const float*)d_in[4];
    const float*     b_ih   = (const float*)d_in[5];
    const float*     b_hh   = (const float*)d_in[6];
    float*           out    = (float*)d_out;

    zero_tail<<<dim3(NN * LL), dim3(64), 0, stream>>>(out, lens);
    rnn_bwd_scan<<<dim3(NN / 16), dim3(512), 0, stream>>>(x, lens, init_s,
                                                          W_ih, W_hh, b_ih, b_hh, out);
}
